// MultiHeadedAttention_v2_76854144795151
// MI455X (gfx1250) — matrix operand order repaired and checked by reading, not compiled, not run on hardware
//
#include <hip/hip_runtime.h>
#include <hip/hip_bf16.h>

// ---------------------------------------------------------------------------
// MI455X (gfx1250) implementation: all GEMM-shaped stages via
// v_wmma_f32_16x16x32_bf16 (wave32), f32 accumulation, inline f32->bf16
// conversion during staging. Workspace (bf16): qw|kw|vw windowed per head,
// yb un-windowed attention output feeding the implicit-GEMM 3x3 conv.
// V tiles in attention phase C staged global->LDS with
// GLOBAL_LOAD_ASYNC_TO_LDS_B128 (ASYNCcnt) when the builtin is available.
// ---------------------------------------------------------------------------

typedef __attribute__((ext_vector_type(16))) __bf16 v16bf;
typedef __attribute__((ext_vector_type(8)))  float  v8f;
typedef __attribute__((ext_vector_type(4)))  int    v4i;

#define AS1 __attribute__((address_space(1)))
#define AS3 __attribute__((address_space(3)))

#if __has_builtin(__builtin_amdgcn_global_load_async_to_lds_b128)
#define HAVE_ASYNC_LDS 1
#endif

#define BT   16
#define Cc   256
#define Hh   96
#define Ww   96
#define HW   (Hh*Ww)        // 9216
#define DKh  64             // d_k per head
#define KCH  32             // WMMA K step (bf16)

__device__ __forceinline__ __bf16 f2bf(float f) {
  unsigned u = __builtin_bit_cast(unsigned, f);
  unsigned r = u + 0x7FFFu + ((u >> 16) & 1u);      // round-to-nearest-even
  unsigned short s = (unsigned short)(r >> 16);
  return __builtin_bit_cast(__bf16, s);
}
__device__ __forceinline__ float bf2f(__bf16 b) {
  unsigned u = ((unsigned)__builtin_bit_cast(unsigned short, b)) << 16;
  return __builtin_bit_cast(float, u);
}

// ---------------------------------------------------------------------------
// Kernel 1: QKV 1x1-conv projection.  Per batch: Y[256 x 9216] = W[256x256]
// times X[256x9216] + bias, output scattered into windowed bf16 layout
// [head][bt][nw][d].  Block tile: M=16 (out-ch) x N=128 (spatial), 8 waves,
// each wave owns one 16x16 N-subtile; K-loop 256/32 = 8 WMMA steps.
// ---------------------------------------------------------------------------
__global__ __launch_bounds__(256)
void qkv_kernel(const float* __restrict__ X, const float* __restrict__ Wm,
                const float* __restrict__ bias, __bf16* __restrict__ out) {
  __shared__ __align__(32) __bf16 ldsA[16 * KCH];    // 16 out-ch x 32 k
  __shared__ __align__(32) __bf16 ldsB[KCH * 128];   // 128 spatial x 32 k

  const int tid = threadIdx.x;
  const int wv = tid >> 5, ln = tid & 31;            // wave32
  const int nt = blockIdx.x % 72, mt = blockIdx.x / 72;
  const int b = blockIdx.y;
  const int o0 = mt * 16, s0 = nt * 128;
  const float* Xb = X + (size_t)b * Cc * HW;

  v8f acc = {};
  for (int kk = 0; kk < Cc; kk += KCH) {
    { // stage A: weights, 512 f32 -> bf16
      int i0 = tid * 2;
      int r = i0 >> 5, c = i0 & 31;
      ldsA[i0]     = f2bf(Wm[(o0 + r) * Cc + kk + c]);
      ldsA[i0 + 1] = f2bf(Wm[(o0 + r) * Cc + kk + c + 1]);
    }
    { // stage B: activations, 32x128 f32 -> bf16 (coalesced float4)
      int rr = tid >> 3;                 // k row 0..31
      int cc0 = (tid & 7) * 16;          // col base
      const float* src = Xb + (size_t)(kk + rr) * HW + s0 + cc0;
      if (kk + KCH < Cc) __builtin_prefetch(src + (size_t)KCH * HW, 0, 1);
      __bf16* dst = ldsB + cc0 * 32 + rr;
#pragma unroll
      for (int u = 0; u < 16; u += 4) {
        float4 f = *(const float4*)(src + u);
        dst[u*32] = f2bf(f.x); dst[(u+1)*32] = f2bf(f.y);
        dst[(u+2)*32] = f2bf(f.z); dst[(u+3)*32] = f2bf(f.w);
      }
    }
    __syncthreads();
    v16bf a  = *(const v16bf*)(ldsA + (ln & 15) * 32 + (ln >> 4) * 16);
    v16bf bb = *(const v16bf*)(ldsB + (wv * 16 + (ln & 15)) * 32 + (ln >> 4) * 16);
    acc = __builtin_amdgcn_wmma_f32_16x16x32_bf16(false, a, false, bb,
                                                  (short)0, acc, false, false);
    __syncthreads();
  }

  // epilogue: bias + scatter into windowed layout
  const int m_hi = (ln >> 4) * 8, ncol = ln & 15;
  const int s = s0 + wv * 16 + ncol;
  const int iy = s / Ww, ix = s % Ww;
#pragma unroll
  for (int r = 0; r < 8; ++r) {
    int o = o0 + r + m_hi;
    float v = acc[r] + bias[o];
    int head = o >> 6, ch = o & 63;
    int p = 2 << head;                                  // 2,4,8,16
    int outw = Ww / p;
    int n = (iy / p) * outw + (ix / p);
    int j = (ch * p + (iy % p)) * p + (ix % p);
    int nw = HW / (p * p), d = DKh * p * p;
    size_t base = (size_t)head * BT * HW * DKh;
    out[base + ((size_t)b * nw + n) * d + j] = f2bf(v);
  }
}

// ---------------------------------------------------------------------------
// Kernel 2: window attention.  One block per (head, batch, 16-query block).
// Scores (bf16) live in dynamic LDS (16 x nwpad <= 74 KB; uses the 320 KB
// WGP LDS).  Phase A: S = scale*Q*K^T via WMMA (operands read directly from
// the contiguous windowed global layout).  Phase B: softmax in LDS.
// Phase C: Y = P*V via WMMA; V tiles staged via async global->LDS DMA.
// ---------------------------------------------------------------------------
__global__ __launch_bounds__(256)
void attn_kernel(const __bf16* __restrict__ qw, const __bf16* __restrict__ kw,
                 const __bf16* __restrict__ vw, __bf16* __restrict__ yb) {
  const int head = blockIdx.z, b = blockIdx.y, qb = blockIdx.x;
  const int p = 2 << head;
  const int d = DKh * p * p;
  const int nw = HW / (p * p);
  const int nqt = (nw + 15) >> 4;
  if (qb >= nqt) return;
  const int nwpad = (nw + 31) & ~31;
  const int nkt = nwpad >> 4;

  extern __shared__ __align__(32) char smem[];
  __bf16* SP = (__bf16*)smem;                     // 16 x nwpad scores/probs
  __shared__ float red[16][16];
  __shared__ float rowmax[16], rowsum[16];
#ifdef HAVE_ASYNC_LDS
  __shared__ __align__(32) __bf16 vtile[8][32 * 16];  // per-wave V tile (1 KB)
#endif

  const int tid = threadIdx.x, wv = tid >> 5, ln = tid & 31;
  const size_t hb = (size_t)head * BT * HW * DKh;
  const __bf16* Q = qw + hb + (size_t)b * nw * d;
  const __bf16* K = kw + hb + (size_t)b * nw * d;
  const __bf16* V = vw + hb + (size_t)b * nw * d;
  const int q0 = qb * 16;
  const float scale = rsqrtf((float)d);
  const int lrow = ln & 15, lsel = (ln >> 4) * 16;
  const int m_hi = (ln >> 4) * 8, ncol = ln & 15;

  // ---- Phase A: scores -----------------------------------------------------
  for (int kt = wv; kt < nkt; kt += 8) {
    v8f acc = {};
    int qrow = q0 + lrow;      if (qrow >= nw) qrow = nw - 1;
    int krow = kt * 16 + lrow; if (krow >= nw) krow = nw - 1;
    const __bf16* qp = Q + (size_t)qrow * d + lsel;
    const __bf16* kp = K + (size_t)krow * d + lsel;
    for (int kc = 0; kc < d; kc += KCH) {
      v16bf a  = *(const v16bf*)(qp + kc);
      v16bf bb = *(const v16bf*)(kp + kc);
      acc = __builtin_amdgcn_wmma_f32_16x16x32_bf16(false, a, false, bb,
                                                    (short)0, acc, false, false);
    }
#pragma unroll
    for (int r = 0; r < 8; ++r) {
      int m = r + m_hi, key = kt * 16 + ncol;
      SP[m * nwpad + key] = f2bf(key < nw ? acc[r] * scale : -3.0e38f);
    }
  }
  __syncthreads();

  // ---- Phase B: softmax (16 threads per row) -------------------------------
  {
    int row = tid >> 4, sub = tid & 15;
    float mx = -3.0e38f;
    for (int c0 = sub; c0 < nwpad; c0 += 16)
      mx = fmaxf(mx, bf2f(SP[row * nwpad + c0]));
    red[row][sub] = mx;
    __syncthreads();
    if (sub == 0) {
      float m = red[row][0];
      for (int i = 1; i < 16; ++i) m = fmaxf(m, red[row][i]);
      rowmax[row] = m;
    }
    __syncthreads();
    float m = rowmax[row], sm = 0.f;
    for (int c0 = sub; c0 < nwpad; c0 += 16)
      sm += __expf(bf2f(SP[row * nwpad + c0]) - m);
    red[row][sub] = sm;
    __syncthreads();
    if (sub == 0) {
      float s = 0.f;
      for (int i = 0; i < 16; ++i) s += red[row][i];
      rowsum[row] = s;
    }
    __syncthreads();
    float inv = 1.0f / rowsum[row];
    for (int c0 = sub; c0 < nwpad; c0 += 16) {
      float e = __expf(bf2f(SP[row * nwpad + c0]) - m) * inv;
      SP[row * nwpad + c0] = f2bf(e);
    }
  }
  __syncthreads();

  // ---- Phase C: Y = P*V, scatter un-windowed -------------------------------
  const int ndt = d >> 4, nk2 = nwpad >> 5;
  const int php2 = p * p, outw = Ww / p;
  for (int dt = wv; dt < ndt; dt += 8) {
    v8f acc = {};
    for (int k2 = 0; k2 < nk2; ++k2) {
      v16bf a = *(const v16bf*)(SP + lrow * nwpad + k2 * 32 + lsel);
      int vrow = k2 * 32 + ln; if (vrow >= nw) vrow = nw - 1;   // P there is 0
      const __bf16* vp = V + (size_t)vrow * d + dt * 16;
#ifdef HAVE_ASYNC_LDS
      // Stage the 32x16 bf16 V tile via async DMA: each lane copies one row
      // as two B128 segments (the immediate offset applies to BOTH global
      // and LDS addresses per ISA 08_async_tensor 4.4).
      asm volatile("s_wait_dscnt 0x0" ::: "memory");  // WAR vs prior ds reads
      __bf16* lp = &vtile[wv][ln * 16];
      __builtin_amdgcn_global_load_async_to_lds_b128(
          (AS1 v4i*)vp, (AS3 v4i*)lp, 0, 0);
      __builtin_amdgcn_global_load_async_to_lds_b128(
          (AS1 v4i*)vp, (AS3 v4i*)lp, 16, 0);
#if __has_builtin(__builtin_amdgcn_s_wait_asynccnt)
      __builtin_amdgcn_s_wait_asynccnt(0);
#else
      asm volatile("s_wait_asynccnt 0x0" ::: "memory");
#endif
      v16bf bb;
      for (int s = 0; s < 16; ++s) bb[s] = vtile[wv][(lsel + s) * 16 + lrow];
#else
      __builtin_prefetch(vp + 16, 0, 1);
      v16bf bb;
      for (int s = 0; s < 16; ++s) {
        int kr = k2 * 32 + lsel + s; if (kr >= nw) kr = nw - 1;
        bb[s] = V[(size_t)kr * d + dt * 16 + lrow];
      }
#endif
      acc = __builtin_amdgcn_wmma_f32_16x16x32_bf16(false, a, false, bb,
                                                    (short)0, acc, false, false);
    }
    int j = dt * 16 + ncol;
    int ch = j / php2, rem = j % php2, py = rem / p, px = rem % p;
#pragma unroll
    for (int r = 0; r < 8; ++r) {
      int n = q0 + r + m_hi;
      if (n < nw) {
        int wy = n / outw, wx = n % outw;
        int iy = wy * p + py, ix = wx * p + px;
        int o = head * DKh + ch;
        yb[(((size_t)b * Cc + o) * Hh + iy) * Ww + ix] = f2bf(acc[r]);
      }
    }
  }
}

// ---------------------------------------------------------------------------
// Kernel 3: 3x3 conv (SAME) as implicit GEMM, K = 256*9 = 2304, fused
// bias + LeakyReLU(0.2), f32 output.  Block tile M=16 x N=128, 8 waves.
// ---------------------------------------------------------------------------
__global__ __launch_bounds__(256)
void conv_kernel(const __bf16* __restrict__ yb, const float* __restrict__ Wo,
                 const float* __restrict__ bo, float* __restrict__ out) {
  __shared__ __align__(32) __bf16 ldsA[16 * KCH];
  __shared__ __align__(32) __bf16 ldsB[KCH * 128];
  const int tid = threadIdx.x, wv = tid >> 5, ln = tid & 31;
  const int nt = blockIdx.x % 72, mt = blockIdx.x / 72;
  const int b = blockIdx.y;
  const int o0 = mt * 16, s0 = nt * 128;
  const __bf16* Yb = yb + (size_t)b * Cc * HW;

  v8f acc = {};
  for (int kk = 0; kk < Cc * 9; kk += KCH) {
    { // stage A: Wo slice (o0..o0+15, kk..kk+31) -> bf16
      int i0 = tid * 2;
#pragma unroll
      for (int u = 0; u < 2; ++u) {
        int i = i0 + u;
        int r = i >> 5, kI = (i & 31) + kk;
        int c = kI / 9, rem = kI % 9;
        ldsA[i] = f2bf(Wo[(((o0 + r) * Cc + c) * 3 + rem / 3) * 3 + rem % 3]);
      }
    }
    { // stage B: im2col gather with zero-pad border
      int rr = tid >> 3;
      int cc0 = (tid & 7) * 16;
      int kI = kk + rr;
      int c = kI / 9, rem = kI % 9, ky = rem / 3 - 1, kx = rem % 3 - 1;
      const __bf16* src = Yb + (size_t)c * HW;
      __builtin_prefetch(src + HW, 0, 1);
#pragma unroll
      for (int u = 0; u < 16; ++u) {
        int s = s0 + cc0 + u;
        int iy = s / Ww + ky, ix = s % Ww + kx;
        __bf16 v = __builtin_bit_cast(__bf16, (unsigned short)0);
        if (iy >= 0 && iy < Hh && ix >= 0 && ix < Ww)
          v = src[iy * Ww + ix];
        ldsB[(cc0 + u) * 32 + rr] = v;
      }
    }
    __syncthreads();
    v16bf a  = *(const v16bf*)(ldsA + (ln & 15) * 32 + (ln >> 4) * 16);
    v16bf bb = *(const v16bf*)(ldsB + (wv * 16 + (ln & 15)) * 32 + (ln >> 4) * 16);
    acc = __builtin_amdgcn_wmma_f32_16x16x32_bf16(false, a, false, bb,
                                                  (short)0, acc, false, false);
    __syncthreads();
  }
  const int m_hi = (ln >> 4) * 8, ncol = ln & 15;
  const int s = s0 + wv * 16 + ncol;
#pragma unroll
  for (int r = 0; r < 8; ++r) {
    int o = o0 + r + m_hi;
    float v = acc[r] + bo[o];
    v = v >= 0.f ? v : 0.2f * v;                      // LeakyReLU(0.2)
    out[((size_t)b * Cc + o) * HW + s] = v;
  }
}

// ---------------------------------------------------------------------------
extern "C" void kernel_launch(void* const* d_in, const int* in_sizes, int n_in,
                              void* d_out, int out_size, void* d_ws, size_t ws_size,
                              hipStream_t stream) {
  const float* x  = (const float*)d_in[0];
  const float* x1 = (const float*)d_in[1];
  const float* Wq = (const float*)d_in[2];
  const float* bq = (const float*)d_in[3];
  const float* Wk = (const float*)d_in[4];
  const float* bk = (const float*)d_in[5];
  const float* Wv = (const float*)d_in[6];
  const float* bv = (const float*)d_in[7];
  const float* Wo = (const float*)d_in[8];
  const float* bo = (const float*)d_in[9];
  float* out = (float*)d_out;

  const size_t TENS = (size_t)BT * Cc * HW;     // 37,748,736 elems
  __bf16* qw = (__bf16*)d_ws;                   // windowed Q  (75.5 MB)
  __bf16* kw = qw + TENS;                       // windowed K
  __bf16* vw = kw + TENS;                       // windowed V
  __bf16* yb = vw + TENS;                       // attention out [bt][c][h][w]

  dim3 blk(256);                                // 8 wave32 waves
  dim3 g1(72 * 16, BT);
  qkv_kernel<<<g1, blk, 0, stream>>>(x,  Wq, bq, qw);
  qkv_kernel<<<g1, blk, 0, stream>>>(x1, Wk, bk, kw);
  qkv_kernel<<<g1, blk, 0, stream>>>(x1, Wv, bv, vw);

  // dynamic LDS: 16 x 2304 bf16 scores for the largest head (73,728 B; WGP
  // LDS is 320 KB so this is fine on gfx1250, but raise the attribute cap).
  const int smem = 16 * 2304 * 2;
  (void)hipFuncSetAttribute((const void*)attn_kernel,
                            hipFuncAttributeMaxDynamicSharedMemorySize, smem);
  dim3 g2(144, BT, 4);
  attn_kernel<<<g2, blk, smem, stream>>>(qw, kw, vw, yb);

  dim3 g3(72 * 16, BT);
  conv_kernel<<<g3, blk, 0, stream>>>(yb, Wo, bo, out);
}
